// MultiHeadAttention_79568564126257
// MI455X (gfx1250) — matrix operand order repaired and checked by reading, not compiled, not run on hardware
//
#include <hip/hip_runtime.h>

// ---------------------------------------------------------------------------
// MHA with seqlen==1 collapses to: out = x @ (Wv@Wp) + (bv@Wp + bp).
//   Wv = w_qkv[:, 4096:6144], bv = b_qkv[4096:6144]
// Kernel 0: x_bf = bf16(x)                          (one pass, v_cvt_pk)
// Kernel 1: bc = bv @ Wp + bp                       (fp32)
// Kernel 2: Wc = bf16( Wv @ Wp )                    (WMMA bf16, fp32 accum)
// Kernel 3: out = x_bf @ Wc + bc                    (WMMA bf16, async-LDS staging)
// ---------------------------------------------------------------------------

typedef __attribute__((ext_vector_type(16))) __bf16 v16bf;
typedef __attribute__((ext_vector_type(2)))  __bf16 v2bf;
typedef __attribute__((ext_vector_type(8)))  float  v8f;
typedef __attribute__((ext_vector_type(4)))  int    v4i;

#define DIM     2048
#define BATCH   16384
#define TILE_M  128
#define TILE_N  128
#define TILE_K  32
#define LDA_S   48            // shorts per A row in LDS (32 data + 16 pad, 96B, 16B-aligned)
#define LDB_S   136           // shorts per B row in LDS (128 data + 8 pad, 272B, 16B-aligned)
#define A_BUF   (TILE_M * LDA_S)   // 6144 shorts
#define B_BUF   (TILE_K * LDB_S)   // 4352 shorts

// ---- fp32 -> packed bf16x2 via native converts ----------------------------
__device__ __forceinline__ unsigned cvt_pk_bf16(float a, float b) {
#if __has_builtin(__builtin_amdgcn_cvt_pk_bf16_f32)
  v2bf p = __builtin_amdgcn_cvt_pk_bf16_f32(a, b);
  return __builtin_bit_cast(unsigned, p);
#else
  v2bf p = {(__bf16)a, (__bf16)b};
  return __builtin_bit_cast(unsigned, p);
#endif
}

__device__ __forceinline__ unsigned short f2bf(float a) {
  __bf16 h = (__bf16)a;
  return __builtin_bit_cast(unsigned short, h);
}

// ---- gfx1250 async global->LDS (ASYNCcnt) with sync fallback --------------
typedef __attribute__((address_space(1))) v4i* gas_v4i_ptr;
typedef __attribute__((address_space(3))) v4i* las_v4i_ptr;

__device__ __forceinline__ void async_copy_b128(const void* g, void* l) {
#if __has_builtin(__builtin_amdgcn_global_load_async_to_lds_b128)
  __builtin_amdgcn_global_load_async_to_lds_b128((gas_v4i_ptr)g, (las_v4i_ptr)l, 0, 0);
#else
  *(uint4*)l = *(const uint4*)g;
#endif
}

__device__ __forceinline__ void wait_async() {
#if __has_builtin(__builtin_amdgcn_global_load_async_to_lds_b128)
#if __has_builtin(__builtin_amdgcn_s_wait_asynccnt)
  __builtin_amdgcn_s_wait_asynccnt(0);
#else
  asm volatile("s_wait_asynccnt 0" ::: "memory");
#endif
#endif
}

union Frag { uint4 q[2]; v16bf v; };

// ---------------------------------------------------------------------------
// Kernel 0: x -> bf16, 8 elements/thread
// ---------------------------------------------------------------------------
__global__ __launch_bounds__(256) void cvt_x_kernel(const float* __restrict__ x,
                                                    unsigned short* __restrict__ xb) {
  size_t i = (size_t)(blockIdx.x * 256 + threadIdx.x) * 8;
  float4 a0 = *(const float4*)(x + i);
  float4 a1 = *(const float4*)(x + i + 4);
  uint4 w;
  w.x = cvt_pk_bf16(a0.x, a0.y);
  w.y = cvt_pk_bf16(a0.z, a0.w);
  w.z = cvt_pk_bf16(a1.x, a1.y);
  w.w = cvt_pk_bf16(a1.z, a1.w);
  *(uint4*)(xb + i) = w;
}

// ---------------------------------------------------------------------------
// Kernel 1: bc[j] = sum_k bv[k] * Wp[k][j] + bp[j]
// ---------------------------------------------------------------------------
__global__ void bias_fold_kernel(const float* __restrict__ b_qkv,
                                 const float* __restrict__ w_proj,
                                 const float* __restrict__ b_proj,
                                 float* __restrict__ bc) {
  int j = blockIdx.x * blockDim.x + threadIdx.x;
  float s = b_proj[j];
  for (int k = 0; k < DIM; ++k)
    s += b_qkv[2 * DIM + k] * w_proj[(size_t)k * DIM + j];
  bc[j] = s;
}

// ---------------------------------------------------------------------------
// 128x128 tile GEMM, 4 waves (wave32), each wave owns 64x64 = 4x4 WMMA tiles.
// BF16_IN=false: A,B fp32 in global, converted to bf16 at staging (Wc GEMM).
// BF16_IN=true : A,B bf16 in global, staged with async global->LDS copies.
// OUT_BF16 selects bf16 store (no bias) vs fp32 store (+bias).
// ---------------------------------------------------------------------------
template <bool BF16_IN, bool OUT_BF16>
__global__ __launch_bounds__(128) void gemm128_bf16_wmma(
    const void* __restrict__ Av, int lda, int acol,
    const void* __restrict__ Bv, int ldb,
    void* __restrict__ Cv, int ldc,
    const float* __restrict__ bias, int K) {

  __shared__ __align__(16) unsigned short sA[2 * A_BUF];
  __shared__ __align__(16) unsigned short sB[2 * B_BUF];

  const int t    = threadIdx.x;
  const int lane = t & 31;
  const int wave = t >> 5;
  const int wm   = (wave >> 1) * 64;   // wave tile origin inside block tile
  const int wn   = (wave & 1)  * 64;
  const int m0g  = blockIdx.y * TILE_M;
  const int n0g  = blockIdx.x * TILE_N;

  auto stage = [&](int k0, int buf) {
    if constexpr (BF16_IN) {
      const unsigned short* Ab = (const unsigned short*)Av;
      const unsigned short* Bb = (const unsigned short*)Bv;
      // A tile: 128 rows x 32 bf16 = 512 x 16B chunks, 4 per thread
#pragma unroll
      for (int q = 0; q < 4; ++q) {
        const int c = t + 128 * q;
        const int row = c >> 2, seg = (c & 3) * 8;
        async_copy_b128(Ab + (size_t)(m0g + row) * lda + acol + k0 + seg,
                        sA + buf * A_BUF + row * LDA_S + seg);
      }
      // B tile: 32 rows x 128 bf16 = 512 x 16B chunks, 4 per thread
#pragma unroll
      for (int q = 0; q < 4; ++q) {
        const int c = t + 128 * q;
        const int row = c >> 4, seg = (c & 15) * 8;
        async_copy_b128(Bb + (size_t)(k0 + row) * ldb + n0g + seg,
                        sB + buf * B_BUF + row * LDB_S + seg);
      }
    } else {
      // fp32 sources, convert with v_cvt while staging
      const float* Af = (const float*)Av;
      const float* Bf = (const float*)Bv;
      {
        const float4* g = (const float4*)(Af + (size_t)(m0g + t) * lda + acol + k0);
        uint4* d = (uint4*)(sA + buf * A_BUF + t * LDA_S);
#pragma unroll
        for (int q = 0; q < 4; ++q) {
          float4 a0 = g[2 * q], a1 = g[2 * q + 1];
          uint4 w;
          w.x = cvt_pk_bf16(a0.x, a0.y); w.y = cvt_pk_bf16(a0.z, a0.w);
          w.z = cvt_pk_bf16(a1.x, a1.y); w.w = cvt_pk_bf16(a1.z, a1.w);
          d[q] = w;
        }
      }
      {
        const int row = t >> 2, col = (t & 3) * 32;
        const float4* g = (const float4*)(Bf + (size_t)(k0 + row) * ldb + n0g + col);
        uint4* d = (uint4*)(sB + buf * B_BUF + row * LDB_S + col);
#pragma unroll
        for (int q = 0; q < 4; ++q) {
          float4 a0 = g[2 * q], a1 = g[2 * q + 1];
          uint4 w;
          w.x = cvt_pk_bf16(a0.x, a0.y); w.y = cvt_pk_bf16(a0.z, a0.w);
          w.z = cvt_pk_bf16(a1.x, a1.y); w.w = cvt_pk_bf16(a1.z, a1.w);
          d[q] = w;
        }
      }
    }
  };

  v8f acc[4][4] = {};

  stage(0, 0);

  // A-fragment lane mapping (ISA 16-bit A 16x32): lane<16 -> row=lane, K in
  // {0..7}U{16..23}; lane>=16 -> row=lane-16, K in {8..15}U{24..31}.
  const int ra = lane & 15;
  const int kb = (lane >> 4) << 3;   // 0 or 8 (in shorts)
  const int nk = K / TILE_K;

  for (int kt = 0; kt < nk; ++kt) {
    if constexpr (BF16_IN) wait_async();
    __syncthreads();
    const int cur = kt & 1;
    if (kt + 1 < nk) stage((kt + 1) * TILE_K, cur ^ 1);

    Frag fa[4], fb[4];
#pragma unroll
    for (int i = 0; i < 4; ++i) {
      const unsigned short* p = sA + cur * A_BUF + (wm + i * 16 + ra) * LDA_S + kb;
      fa[i].q[0] = *(const uint4*)(p);        // K kb..kb+7
      fa[i].q[1] = *(const uint4*)(p + 16);   // K kb+16..kb+23
    }
    // B-fragment lane mapping (ISA 16-bit B 32x16): lane holds column N=lane%16,
    // K in {kb..kb+7}U{kb+16..kb+23} (same order as A).
#pragma unroll
    for (int j = 0; j < 4; ++j) {
      const unsigned short* p = sB + cur * B_BUF + kb * LDB_S + wn + j * 16 + ra;
#pragma unroll
      for (int s = 0; s < 8; ++s) {
        fb[j].v[s]     = __builtin_bit_cast(__bf16, p[s * LDB_S]);
        fb[j].v[s + 8] = __builtin_bit_cast(__bf16, p[(s + 16) * LDB_S]);
      }
    }

#pragma unroll
    for (int i = 0; i < 4; ++i)
#pragma unroll
      for (int j = 0; j < 4; ++j)
        acc[i][j] = __builtin_amdgcn_wmma_f32_16x16x32_bf16(
            false, fa[i].v, false, fb[j].v, (short)0, acc[i][j], false, false);
  }

  // C/D layout: VGPR r -> (lane<16: M=r, N=lane) / (lane>=16: M=8+r, N=lane-16)
  const int colb = lane & 15;
  const int rowb = (lane >> 4) * 8;
#pragma unroll
  for (int i = 0; i < 4; ++i) {
#pragma unroll
    for (int j = 0; j < 4; ++j) {
      const int col   = n0g + wn + j * 16 + colb;
      const int rbase = m0g + wm + i * 16 + rowb;
      if constexpr (OUT_BF16) {
        unsigned short* C = (unsigned short*)Cv;
#pragma unroll
        for (int r = 0; r < 8; ++r)
          C[(size_t)(rbase + r) * ldc + col] = f2bf(acc[i][j][r]);
      } else {
        float* C = (float*)Cv;
        const float bb = bias[col];
#pragma unroll
        for (int r = 0; r < 8; ++r)
          C[(size_t)(rbase + r) * ldc + col] = acc[i][j][r] + bb;
      }
    }
  }
}

// ---------------------------------------------------------------------------
extern "C" void kernel_launch(void* const* d_in, const int* in_sizes, int n_in,
                              void* d_out, int out_size, void* d_ws, size_t ws_size,
                              hipStream_t stream) {
  const float* x      = (const float*)d_in[0];   // [16384, 2048]
  const float* w_qkv  = (const float*)d_in[1];   // [2048, 6144]
  const float* b_qkv  = (const float*)d_in[2];   // [6144]
  const float* w_proj = (const float*)d_in[3];   // [2048, 2048]
  const float* b_proj = (const float*)d_in[4];   // [2048]
  float* out = (float*)d_out;                    // [16384, 2048]

  // workspace: bc (2048 f32) @0 | Wc (2048*2048 bf16) @16KB | x_bf @16KB+8MB
  float* bc = (float*)d_ws;
  unsigned short* wc = (unsigned short*)((char*)d_ws + 16384);
  unsigned short* xb = (unsigned short*)((char*)d_ws + 16384 + (size_t)DIM * DIM * 2);

  // 0) x -> bf16
  cvt_x_kernel<<<(BATCH * DIM) / (256 * 8), 256, 0, stream>>>(x, xb);

  // 1) folded bias
  bias_fold_kernel<<<DIM / 256, 256, 0, stream>>>(b_qkv, w_proj, b_proj, bc);

  // 2) Wc = bf16( Wv @ Wp );  Wv = w_qkv[:, 4096:6144] (row stride 3*DIM)
  gemm128_bf16_wmma<false, true><<<dim3(DIM / TILE_N, DIM / TILE_M), 128, 0, stream>>>(
      (const void*)w_qkv, 3 * DIM, 2 * DIM, (const void*)w_proj, DIM,
      (void*)wc, DIM, nullptr, DIM);

  // 3) out = x_bf @ Wc + bc   (async global->LDS staging)
  gemm128_bf16_wmma<true, false><<<dim3(DIM / TILE_N, BATCH / TILE_M), 128, 0, stream>>>(
      (const void*)xb, DIM, 0, (const void*)wc, DIM,
      (void*)out, DIM, bc, DIM);
}
